// Model_21251498181126
// MI455X (gfx1250) — matrix operand order repaired and checked by reading, not compiled, not run on hardware
//
#include <hip/hip_runtime.h>

// ---------------------------------------------------------------------------
// Types for WMMA bf16 (gfx1250: V_WMMA_F32_16X16X32_BF16)
// ---------------------------------------------------------------------------
typedef __attribute__((ext_vector_type(16))) __bf16 bf16x16;
typedef __attribute__((ext_vector_type(8)))  float  f32x8;

union Frag {
    unsigned int u[8];
    uint4        q[2];
    bf16x16      v;
};

__device__ __forceinline__ unsigned short f2bf(float f) {
    unsigned int u = __float_as_uint(f);
    unsigned int r = (u + 0x7FFFu + ((u >> 16) & 1u)) >> 16;   // RNE
    return (unsigned short)r;
}
__device__ __forceinline__ unsigned int pk2bf(float lo, float hi) {
    return (unsigned)f2bf(lo) | ((unsigned)f2bf(hi) << 16);
}
__device__ __forceinline__ float bf2f(unsigned int bits16) {
    return __uint_as_float(bits16 << 16);
}
__device__ __forceinline__ float sigf(float x) {
    return 1.0f / (1.0f + __expf(-x));
}

// Problem constants
#define TT   2048
#define BB   64
#define ROWS (TT * BB)      // 131072
#define GP   1216           // gate-padded width: 4 gates x 304
#define HPAD 304            // hidden 300 padded to 19 tiles of 16
#define HLD  320            // hidden K padded to 10 k-tiles of 32
#define NJT  19             // hidden n-tiles
#define NTG  76             // 4 gates * 19
#define LROWS 1024          // leaf rows (16*64)

// ---------------------------------------------------------------------------
// Weight composition: out[i][n] = sum_e E[i][e] * W[e*N + n]
// ---------------------------------------------------------------------------
__global__ void compose_w(const float* __restrict__ E, const float* __restrict__ W,
                          int I, int Ed, int N, float* __restrict__ out) {
    int idx = blockIdx.x * blockDim.x + threadIdx.x;
    if (idx >= I * N) return;
    int i = idx / N, n = idx % N;
    float s = 0.f;
    for (int e = 0; e < Ed; e++) s += E[i * Ed + e] * W[e * N + n];
    out[idx] = s;
}

// Gate-padded bias (1216): bias[g*304+j] = emb_b . W_ih[:,g*300+j] + gate_b
__global__ void compose_bias(const float* __restrict__ eb, const float* __restrict__ W,
                             const float* __restrict__ gb, int Ed, float* __restrict__ out) {
    int np = blockIdx.x * blockDim.x + threadIdx.x;
    if (np >= GP) return;
    int g = np / HPAD, j = np % HPAD;
    float s = 0.f;
    if (j < 300) {
        int n = g * 300 + j;
        s = gb[n];
        for (int e = 0; e < Ed; e++) s += eb[e] * W[e * 1200 + n];
    }
    out[np] = s;
}

// ---------------------------------------------------------------------------
// Pack f32 weight matrix (K x N, row stride ldw) into WMMA bf16 B-fragment
// tiles.  Tile (nt,kt) = 256 dwords: lane-major, 8 dwords per lane.
// ---------------------------------------------------------------------------
__global__ void pack_b(const float* __restrict__ W, int ldw, int Kreal, int Nreal,
                       int gatePad, int KT, int NT, unsigned int* __restrict__ out) {
    int idx = blockIdx.x * blockDim.x + threadIdx.x;
    int total = KT * NT * 256;
    if (idx >= total) return;
    int tile = idx >> 8, w = idx & 255;
    int lane = w >> 3, r = w & 7;
    int nt = tile / KT, kt = tile % KT;
    int koff = (lane < 16) ? 0 : 8;
    int k0 = kt * 32 + koff + 2 * r + ((r & 4) ? 8 : 0);
    int npk = nt * 16 + (lane & 15);
    int ncol; bool nok;
    if (gatePad) { int g = npk / HPAD, j = npk % HPAD; ncol = g * 300 + j; nok = (j < 300); }
    else         { ncol = npk; nok = (npk < Nreal); }
    float f0 = 0.f, f1 = 0.f;
    if (nok) {
        if (k0     < Kreal) f0 = W[(size_t)k0 * ldw + ncol];
        if (k0 + 1 < Kreal) f1 = W[(size_t)(k0 + 1) * ldw + ncol];
    }
    out[idx] = pk2bf(f0, f1);
}

__global__ void fill_zero_u32(unsigned int* __restrict__ p, long long n) {
    long long i = (long long)blockIdx.x * blockDim.x + threadIdx.x;
    if (i < n) p[i] = 0u;
}

// ---------------------------------------------------------------------------
// WMMA GEMM: TWO adjacent 16x16 C tiles per wave (A reused, two independent
// WMMA chains).  K fully padded (KT*32), no K guards; straight-line k-loop.
//   ABF16:  A is bf16 row-major (else f32 row-major, cvt on load; rows with
//           index < rowShift read as zero)
//   CBF16:  C stored as bf16 (else f32)
//   CTRANS: C stored transposed [col][row] (ldc = #rows); contiguous 8-row
//           vector stores per lane (feeds the sequential LSTM's gate init)
// ---------------------------------------------------------------------------
template <bool ABF16, bool CBF16, bool CTRANS>
__global__ __launch_bounds__(256) void gemm_wmma(
    const float* __restrict__ Af32, const unsigned short* __restrict__ Abf16,
    int lda, int rowShift, int mtiles, int KT,
    const unsigned int* __restrict__ Bpack, int NT,
    const float* __restrict__ bias,
    float* __restrict__ Cf32, unsigned short* __restrict__ Cbf16, int ldc) {

    const int NT2 = NT >> 1;
    int wid  = (blockIdx.x * blockDim.x + threadIdx.x) >> 5;
    int lane = threadIdx.x & 31;
    if (wid >= mtiles * NT2) return;
    int mt = wid / NT2, np = wid % NT2;
    int nt0 = np * 2, nt1 = nt0 + 1;
    int m0 = mt * 16;
    int arow  = m0 + (lane & 15);
    int koffA = (lane < 16) ? 0 : 8;

    f32x8 acc0, acc1;
#pragma unroll
    for (int e = 0; e < 8; e++) { acc0[e] = 0.f; acc1[e] = 0.f; }

    int srow = arow - rowShift;
    int crow = srow < 0 ? 0 : srow;
    const float* ap = ABF16 ? nullptr : (Af32 + (size_t)crow * lda + koffA);
    const unsigned short* ab = ABF16 ? (Abf16 + (size_t)arow * lda + koffA) : nullptr;
    bool zr = (srow < 0);

    const unsigned int* bt0 = Bpack + (((size_t)nt0 * KT) << 8) + (lane << 3);
    const unsigned int* bt1 = bt0 + ((size_t)KT << 8);

    for (int kt = 0; kt < KT; kt++) {
        Frag a;
        if (ABF16) {
            a.q[0] = *(const uint4*)(ab);
            a.q[1] = *(const uint4*)(ab + 16);
            ab += 32;
        } else {
            float4 f0 = *(const float4*)(ap);
            float4 f1 = *(const float4*)(ap + 4);
            float4 f2 = *(const float4*)(ap + 16);
            float4 f3 = *(const float4*)(ap + 20);
            ap += 32;
            a.u[0] = pk2bf(f0.x, f0.y);
            a.u[1] = pk2bf(f0.z, f0.w);
            a.u[2] = pk2bf(f1.x, f1.y);
            a.u[3] = pk2bf(f1.z, f1.w);
            a.u[4] = pk2bf(f2.x, f2.y);
            a.u[5] = pk2bf(f2.z, f2.w);
            a.u[6] = pk2bf(f3.x, f3.y);
            a.u[7] = pk2bf(f3.z, f3.w);
            if (zr) {
#pragma unroll
                for (int d = 0; d < 8; d++) a.u[d] = 0u;
            }
        }
        Frag b0, b1;
        b0.q[0] = *(const uint4*)(bt0);
        b0.q[1] = *(const uint4*)(bt0 + 4);
        b1.q[0] = *(const uint4*)(bt1);
        b1.q[1] = *(const uint4*)(bt1 + 4);
        bt0 += 256; bt1 += 256;
        acc0 = __builtin_amdgcn_wmma_f32_16x16x32_bf16(false, a.v, false, b0.v,
                                                       (short)0, acc0, false, false);
        acc1 = __builtin_amdgcn_wmma_f32_16x16x32_bf16(false, a.v, false, b1.v,
                                                       (short)0, acc1, false, false);
    }

    int lcol  = lane & 15;
    int rbase = m0 + 8 * (lane >> 4);
    int col0 = nt0 * 16 + lcol, col1 = nt1 * 16 + lcol;
    float bv0 = bias ? bias[col0] : 0.f;
    float bv1 = bias ? bias[col1] : 0.f;

    if (CTRANS) {
        // [col][row] layout; lane's 8 rows are contiguous -> vector stores
        if (CBF16) {
            uint4 w0, w1;
            w0.x = pk2bf(acc0[0] + bv0, acc0[1] + bv0);
            w0.y = pk2bf(acc0[2] + bv0, acc0[3] + bv0);
            w0.z = pk2bf(acc0[4] + bv0, acc0[5] + bv0);
            w0.w = pk2bf(acc0[6] + bv0, acc0[7] + bv0);
            w1.x = pk2bf(acc1[0] + bv1, acc1[1] + bv1);
            w1.y = pk2bf(acc1[2] + bv1, acc1[3] + bv1);
            w1.z = pk2bf(acc1[4] + bv1, acc1[5] + bv1);
            w1.w = pk2bf(acc1[6] + bv1, acc1[7] + bv1);
            *(uint4*)(Cbf16 + (size_t)col0 * ldc + rbase) = w0;
            *(uint4*)(Cbf16 + (size_t)col1 * ldc + rbase) = w1;
        } else {
            float4 w0a = make_float4(acc0[0] + bv0, acc0[1] + bv0, acc0[2] + bv0, acc0[3] + bv0);
            float4 w0b = make_float4(acc0[4] + bv0, acc0[5] + bv0, acc0[6] + bv0, acc0[7] + bv0);
            float4 w1a = make_float4(acc1[0] + bv1, acc1[1] + bv1, acc1[2] + bv1, acc1[3] + bv1);
            float4 w1b = make_float4(acc1[4] + bv1, acc1[5] + bv1, acc1[6] + bv1, acc1[7] + bv1);
            float* p0 = Cf32 + (size_t)col0 * ldc + rbase;
            float* p1 = Cf32 + (size_t)col1 * ldc + rbase;
            *(float4*)(p0) = w0a; *(float4*)(p0 + 4) = w0b;
            *(float4*)(p1) = w1a; *(float4*)(p1 + 4) = w1b;
        }
    } else {
#pragma unroll
        for (int e = 0; e < 8; e++) {
            float v0 = acc0[e] + bv0;
            float v1 = acc1[e] + bv1;
            size_t o0 = (size_t)(rbase + e) * ldc + col0;
            size_t o1 = (size_t)(rbase + e) * ldc + col1;
            if (CBF16) { Cbf16[o0] = f2bf(v0); Cbf16[o1] = f2bf(v1); }
            else       { Cf32[o0]  = v0;       Cf32[o1]  = v1; }
        }
    }
}

// ---------------------------------------------------------------------------
// Persistent sequential LSTM: one workgroup, 32 waves, h in LDS (bf16),
// c in registers.  gates = pre[t] (+leafG[t>>7]) + h @ Whh (WMMA bf16).
// pre and leafG are TRANSPOSED [col][row] so the gate init is one b128 load
// per gate.  Cell grid: 4 m-tiles x 19 hidden n-tiles = 76 cells; wave w
// owns cells {w, w+32, w+64}; 4 independent gate WMMA chains share each A.
// ---------------------------------------------------------------------------
template <bool HASLEAF, bool ISENC>
__global__ __launch_bounds__(1024) void lstm_seq(
    const unsigned short* __restrict__ pre,   // [1216][T*64] bf16 (transposed)
    const float* __restrict__ leafG,          // HASLEAF: [1216][1024] f32 (transposed)
    const unsigned int* __restrict__ Whh,     // 76 x 10 packed tiles
    float* __restrict__ esc,                  // ISENC: [16][64][304] f32
    unsigned short* __restrict__ hsOut,       // !ISENC: [T*64][320] bf16
    int T) {

    __shared__ unsigned short hbuf[BB * HLD];           // 40 KB
    int tid = threadIdx.x;
    for (int i = tid; i < BB * HLD / 2; i += 1024) ((unsigned int*)hbuf)[i] = 0u;
    __syncthreads();

    int wave = tid >> 5, lane = tid & 31;
    int lrow = lane & 15, lhalf = lane >> 4;
    int koffA = (lane < 16) ? 0 : 8;

    int cells[3], mt[3], jt[3];
#pragma unroll
    for (int s = 0; s < 3; s++) {
        cells[s] = wave + 32 * s;
        mt[s] = cells[s] / NJT;
        jt[s] = cells[s] % NJT;
    }
    float cst[3][8];
#pragma unroll
    for (int s = 0; s < 3; s++)
#pragma unroll
        for (int e = 0; e < 8; e++) cst[s][e] = 0.f;
    unsigned short hres[3][8];

    for (int t = 0; t < T; t++) {
#pragma unroll
        for (int s = 0; s < 3; s++) {
            if (cells[s] >= NTG) continue;
            int m0 = mt[s] * 16, n0 = jt[s] * 16;
            int col = n0 + lrow;
            int rbase = m0 + 8 * lhalf;

            f32x8 acc[4];
#pragma unroll
            for (int g = 0; g < 4; g++) {
                // one vector load: 8 consecutive rows of transposed pre
                const unsigned short* pg =
                    pre + (size_t)(g * HPAD + col) * ROWS + t * BB + rbase;
                uint4 q = *(const uint4*)pg;
                acc[g][0] = bf2f(q.x & 0xffffu); acc[g][1] = bf2f(q.x >> 16);
                acc[g][2] = bf2f(q.y & 0xffffu); acc[g][3] = bf2f(q.y >> 16);
                acc[g][4] = bf2f(q.z & 0xffffu); acc[g][5] = bf2f(q.z >> 16);
                acc[g][6] = bf2f(q.w & 0xffffu); acc[g][7] = bf2f(q.w >> 16);
                if (HASLEAF) {
                    const float* lg =
                        leafG + (size_t)(g * HPAD + col) * LROWS + (t >> 7) * BB + rbase;
                    float4 l0 = *(const float4*)(lg);
                    float4 l1 = *(const float4*)(lg + 4);
                    acc[g][0] += l0.x; acc[g][1] += l0.y;
                    acc[g][2] += l0.z; acc[g][3] += l0.w;
                    acc[g][4] += l1.x; acc[g][5] += l1.y;
                    acc[g][6] += l1.z; acc[g][7] += l1.w;
                }
            }

            int arow = m0 + lrow;
            const unsigned short* hr = hbuf + arow * HLD + koffA;
            const unsigned int* btg[4];
#pragma unroll
            for (int g = 0; g < 4; g++)
                btg[g] = Whh + (((size_t)((g * NJT + jt[s]) * 10)) << 8) + (lane << 3);

            for (int kt = 0; kt < 10; kt++) {
                Frag a;
                a.q[0] = *(const uint4*)(hr);
                a.q[1] = *(const uint4*)(hr + 16);
                hr += 32;
#pragma unroll
                for (int g = 0; g < 4; g++) {
                    Frag b;
                    b.q[0] = *(const uint4*)(btg[g]);
                    b.q[1] = *(const uint4*)(btg[g] + 4);
                    btg[g] += 256;
                    acc[g] = __builtin_amdgcn_wmma_f32_16x16x32_bf16(
                        false, a.v, false, b.v, (short)0, acc[g], false, false);
                }
            }

#pragma unroll
            for (int e = 0; e < 8; e++) {
                float ig = sigf(acc[0][e]);
                float fg = sigf(acc[1][e]);
                float gg = tanhf(acc[2][e]);
                float og = sigf(acc[3][e]);
                float cv = fg * cst[s][e] + ig * gg;
                cst[s][e] = cv;
                hres[s][e] = f2bf(og * tanhf(cv));
            }
        }
        __syncthreads();   // all reads of old h complete
#pragma unroll
        for (int s = 0; s < 3; s++) {
            if (cells[s] >= NTG) continue;
            int col = jt[s] * 16 + lrow;
            int rbase = mt[s] * 16 + 8 * lhalf;
#pragma unroll
            for (int e = 0; e < 8; e++) {
                int row = rbase + e;
                unsigned short hb = hres[s][e];
                hbuf[row * HLD + col] = hb;
                if (!ISENC) hsOut[(size_t)(t * BB + row) * HLD + col] = hb;
                if (ISENC && ((t & 127) == 127))
                    esc[((t >> 7) * BB + row) * HPAD + col] = bf2f((unsigned)hb);
            }
        }
        __syncthreads();   // new h visible
    }
}

// ---------------------------------------------------------------------------
// Scalar tree/MLP kernels (tiny FLOP counts; single block each)
// ---------------------------------------------------------------------------
__global__ __launch_bounds__(1024) void reduce_scan(
    const float* __restrict__ esc, const float* __restrict__ rw,
    const float* __restrict__ rb, float* __restrict__ code) {
    extern __shared__ float sm[];
    float* cur = sm;
    float* nxt = sm + 19200;
    int tid = threadIdx.x;
    for (int idx = tid; idx < 19200; idx += 1024) {
        int b = idx / 300, n = idx % 300;
        cur[idx] = esc[b * HPAD + n];
    }
    __syncthreads();
    for (int s = 1; s < 16; s++) {
        const float* ev = esc + s * BB * HPAD;
        for (int idx = tid; idx < 19200; idx += 1024) {
            int b = idx / 300, n = idx % 300;
            float sum = rb[n];
            const float* cb = cur + b * 300;
            const float* eb = ev + b * HPAD;
            for (int k = 0; k < 300; k++) sum += cb[k] * rw[k * 300 + n];
            for (int k = 0; k < 300; k++) sum += eb[k] * rw[(300 + k) * 300 + n];
            nxt[idx] = fmaxf(sum, 0.f);
        }
        __syncthreads();
        float* tw = cur; cur = nxt; nxt = tw;
    }
    for (int idx = tid; idx < 19200; idx += 1024) code[idx] = cur[idx];
}

__global__ __launch_bounds__(1024) void vae_kernel(
    const float* __restrict__ code,
    const float* vm0w, const float* vm0b, const float* vm1w, const float* vm1b,
    const float* vs0w, const float* vs0b, const float* vs1w, const float* vs1b,
    float* __restrict__ mu_out, float* __restrict__ std_out) {
    extern __shared__ float sm[];
    float* t1 = sm;
    float* t2 = sm + 12800;
    int tid = threadIdx.x;
    for (int idx = tid; idx < 12800; idx += 1024) {
        int b = idx / 200, j = idx % 200;
        float s1 = vm0b[j], s2 = vs0b[j];
        const float* cb = code + b * 300;
        for (int k = 0; k < 300; k++) {
            float c = cb[k];
            s1 += c * vm0w[k * 200 + j];
            s2 += c * vs0w[k * 200 + j];
        }
        t1[idx] = fmaxf(s1, 0.f);
        t2[idx] = fmaxf(s2, 0.f);
    }
    __syncthreads();
    for (int idx = tid; idx < 19200; idx += 1024) {
        int b = idx / 300, n = idx % 300;
        float s1 = vm1b[n], s2 = vs1b[n];
        const float* a1 = t1 + b * 200;
        const float* a2 = t2 + b * 200;
        for (int k = 0; k < 200; k++) {
            s1 += a1[k] * vm1w[k * 300 + n];
            s2 += a2[k] * vs1w[k * 300 + n];
        }
        mu_out[idx]  = s1;
        std_out[idx] = (s2 > 20.f) ? s2 : log1pf(__expf(s2));
    }
}

// leaves written with row stride 320 (K-padded for the leafG WMMA GEMM)
__global__ __launch_bounds__(1024) void split_kernel(
    const float* __restrict__ z,
    const float* cl0w, const float* cl0b, const float* cl1w, const float* cl1b,
    const float* splw, const float* splb,
    float* __restrict__ cls_out, float* __restrict__ leaves) {
    extern __shared__ float sm[];
    float* cur = sm;
    float* nxt = sm + 19200;
    float* tmp = sm + 38400;
    int tid = threadIdx.x;
    for (int idx = tid; idx < 19200; idx += 1024) cur[idx] = z[idx];
    for (int idx = tid; idx < 1024 * 20; idx += 1024) {
        int row = idx / 20, c = idx % 20;
        leaves[(size_t)row * HLD + 300 + c] = 0.f;      // zero K padding
    }
    __syncthreads();
    for (int s = 0; s < 15; s++) {
        for (int idx = tid; idx < 3200; idx += 1024) {
            int b = idx / 50, j = idx % 50;
            float t = cl0b[j];
            const float* cb = cur + b * 300;
            for (int k = 0; k < 300; k++) t += cb[k] * cl0w[k * 50 + j];
            tmp[idx] = fmaxf(t, 0.f);
        }
        for (int idx = tid; idx < 38400; idx += 1024) {
            int b = idx / 600, n = idx % 600;
            float v = splb[n];
            const float* cb = cur + b * 300;
            for (int k = 0; k < 300; k++) v += cb[k] * splw[k * 600 + n];
            if (n < 300) nxt[b * 300 + n] = v;
            else         leaves[(size_t)((15 - s) * BB + b) * HLD + (n - 300)] = v;
        }
        __syncthreads();
        for (int idx = tid; idx < 128; idx += 1024) {
            int b = idx / 2, j = idx % 2;
            float v = cl1b[j];
            const float* tb = tmp + b * 50;
            for (int k = 0; k < 50; k++) v += tb[k] * cl1w[k * 2 + j];
            cls_out[s * 128 + idx] = v;
        }
        __syncthreads();
        float* tw = cur; cur = nxt; nxt = tw;
    }
    for (int idx = tid; idx < 19200; idx += 1024) {
        int b = idx / 300, n = idx % 300;
        leaves[(size_t)b * HLD + n] = cur[idx];
    }
}

__global__ __launch_bounds__(512) void length_kernel(
    const float* __restrict__ z, const float* ld0w, const float* ld0b,
    const float* ld1w, const float* ld1b, float* __restrict__ out) {
    extern __shared__ float sm[];
    int tid = threadIdx.x;
    for (int idx = tid; idx < 3200; idx += 512) {
        int b = idx / 50, j = idx % 50;
        float t = ld0b[j];
        const float* zb = z + b * 300;
        for (int k = 0; k < 300; k++) t += zb[k] * ld0w[k * 50 + j];
        sm[idx] = fmaxf(t, 0.f);
    }
    __syncthreads();
    for (int idx = tid; idx < 32000; idx += 512) {
        int b = idx / 500, n = idx % 500;
        float v = ld1b[n];
        const float* tb = sm + b * 50;
        for (int k = 0; k < 50; k++) v += tb[k] * ld1w[k * 500 + n];
        out[idx] = v;
    }
}

// ---------------------------------------------------------------------------
// Host launch
// ---------------------------------------------------------------------------
extern "C" void kernel_launch(void* const* d_in, const int* in_sizes, int n_in,
                              void* d_out, int out_size, void* d_ws, size_t ws_size,
                              hipStream_t stream) {
    const float* x         = (const float*)d_in[0];
    const float* enc_emb_w = (const float*)d_in[1];
    const float* enc_emb_b = (const float*)d_in[2];
    const float* enc_ih_w  = (const float*)d_in[3];
    const float* enc_hh_w  = (const float*)d_in[4];
    const float* enc_b     = (const float*)d_in[5];
    const float* red_w     = (const float*)d_in[6];
    const float* red_b     = (const float*)d_in[7];
    const float* vm0_w     = (const float*)d_in[8];
    const float* vm0_b     = (const float*)d_in[9];
    const float* vm1_w     = (const float*)d_in[10];
    const float* vm1_b     = (const float*)d_in[11];
    const float* vs0_w     = (const float*)d_in[12];
    const float* vs0_b     = (const float*)d_in[13];
    const float* vs1_w     = (const float*)d_in[14];
    const float* vs1_b     = (const float*)d_in[15];
    const float* spl_w     = (const float*)d_in[16];
    const float* spl_b     = (const float*)d_in[17];
    const float* cl0_w     = (const float*)d_in[18];
    const float* cl0_b     = (const float*)d_in[19];
    const float* cl1_w     = (const float*)d_in[20];
    const float* cl1_b     = (const float*)d_in[21];
    const float* ld0_w     = (const float*)d_in[22];
    const float* ld0_b     = (const float*)d_in[23];
    const float* ld1_w     = (const float*)d_in[24];
    const float* ld1_b     = (const float*)d_in[25];
    const float* dec_emb_w = (const float*)d_in[26];
    const float* dec_emb_b = (const float*)d_in[27];
    const float* dec_ih_w  = (const float*)d_in[28];
    const float* dec_hh_w  = (const float*)d_in[29];
    const float* dec_b     = (const float*)d_in[30];
    const float* dec_out_w = (const float*)d_in[31];
    const float* dec_out_b = (const float*)d_in[32];

    float* dout = (float*)d_out;
    const size_t OUT_N   = (size_t)ROWS * 384;          // 50331648
    float* cls_out = dout + OUT_N;                      // 15*64*2
    float* len_out = cls_out + 15 * 64 * 2;             // 64*500
    float* mu_out  = len_out + 64 * 500;                // 64*300
    float* std_out = mu_out + 19200;                    // 64*300

    // workspace bump allocator
    char* ws = (char*)d_ws;
    size_t off = 0;
    auto alloc = [&](size_t bytes) -> void* {
        void* p = ws + off;
        off += (bytes + 255) & ~(size_t)255;
        return p;
    };
    unsigned short* preEnc = (unsigned short*)alloc((size_t)ROWS * GP * 2);  // transposed [GP][ROWS]
    unsigned short* preDec = (unsigned short*)alloc((size_t)ROWS * GP * 2);  // transposed
    unsigned short* hsD    = (unsigned short*)alloc((size_t)ROWS * HLD * 2); // row-major [ROWS][320]
    unsigned int* pkEncIh  = (unsigned int*)alloc((size_t)4  * NTG * 1024);
    unsigned int* pkDecIh  = (unsigned int*)alloc((size_t)4  * NTG * 1024);
    unsigned int* pkEncHh  = (unsigned int*)alloc((size_t)10 * NTG * 1024);
    unsigned int* pkDecHh  = (unsigned int*)alloc((size_t)10 * NTG * 1024);
    unsigned int* pkLeafW  = (unsigned int*)alloc((size_t)10 * NTG * 1024);
    unsigned int* pkOutW   = (unsigned int*)alloc((size_t)10 * 24  * 1024);
    float* encW    = (float*)alloc((size_t)128 * 1200 * 4);
    float* decW    = (float*)alloc((size_t)128 * 1200 * 4);
    float* encBias = (float*)alloc(GP * 4);
    float* decBias = (float*)alloc(GP * 4);
    float* escBuf  = (float*)alloc((size_t)16 * BB * HPAD * 4);
    float* codeBuf = (float*)alloc(19200 * 4);
    float* leaves  = (float*)alloc((size_t)LROWS * HLD * 4);     // K-padded to 320
    float* leafG   = (float*)alloc((size_t)GP * LROWS * 4);      // transposed [GP][1024]

    // 1) compose input-side affine weights (emb @ ih) + biases
    {
        int n = 128 * 1200;
        compose_w<<<(n + 255) / 256, 256, 0, stream>>>(enc_emb_w, enc_ih_w, 128, 150, 1200, encW);
        compose_w<<<(n + 255) / 256, 256, 0, stream>>>(dec_emb_w, dec_ih_w, 128, 150, 1200, decW);
        compose_bias<<<(GP + 255) / 256, 256, 0, stream>>>(enc_emb_b, enc_ih_w, enc_b, 150, encBias);
        compose_bias<<<(GP + 255) / 256, 256, 0, stream>>>(dec_emb_b, dec_ih_w, dec_b, 150, decBias);
    }
    // 2) pack all B matrices into WMMA fragment layout
    {
        int n4  = 4  * NTG * 256;
        int n10 = 10 * NTG * 256;
        int n24 = 10 * 24  * 256;
        pack_b<<<(n4  + 255) / 256, 256, 0, stream>>>(encW, 1200, 128, 0, 1, 4,  NTG, pkEncIh);
        pack_b<<<(n4  + 255) / 256, 256, 0, stream>>>(decW, 1200, 128, 0, 1, 4,  NTG, pkDecIh);
        pack_b<<<(n10 + 255) / 256, 256, 0, stream>>>(enc_hh_w, 1200, 300, 0, 1, 10, NTG, pkEncHh);
        pack_b<<<(n10 + 255) / 256, 256, 0, stream>>>(dec_hh_w, 1200, 300, 0, 1, 10, NTG, pkDecHh);
        pack_b<<<(n10 + 255) / 256, 256, 0, stream>>>(dec_ih_w + 150 * 1200, 1200, 300, 0, 1, 10, NTG, pkLeafW);
        pack_b<<<(n24 + 255) / 256, 256, 0, stream>>>(dec_out_w, 384, 300, 384, 0, 10, 24, pkOutW);
    }
    // 3) zero hsD (padding columns must be clean)
    {
        long long n = (long long)ROWS * HLD / 2;
        fill_zero_u32<<<(unsigned)((n + 255) / 256), 256, 0, stream>>>((unsigned int*)hsD, n);
    }
    // 4) big parallel input-gate GEMMs (x -> transposed pre-gates, bf16)
    {
        int tiles = (ROWS / 16) * (NTG / 2);            // 311296 waves
        int blocks = (tiles * 32 + 255) / 256;
        gemm_wmma<false, true, true><<<blocks, 256, 0, stream>>>(
            x, nullptr, 128, 0,  ROWS / 16, 4, pkEncIh, NTG, encBias, nullptr, preEnc, ROWS);
        gemm_wmma<false, true, true><<<blocks, 256, 0, stream>>>(
            x, nullptr, 128, BB, ROWS / 16, 4, pkDecIh, NTG, decBias, nullptr, preDec, ROWS);
    }
    // 5) encoder recurrence (persistent workgroup, WMMA on h@Whh)
    lstm_seq<false, true><<<1, 1024, 0, stream>>>(preEnc, nullptr, pkEncHh, escBuf, nullptr, TT);

    // 6) tree / MLP chain
    reduce_scan<<<1, 1024, 2 * 19200 * 4, stream>>>(escBuf, red_w, red_b, codeBuf);
    vae_kernel<<<1, 1024, 2 * 12800 * 4, stream>>>(codeBuf,
        vm0_w, vm0_b, vm1_w, vm1_b, vs0_w, vs0_b, vs1_w, vs1_b, mu_out, std_out);
    split_kernel<<<1, 1024, (2 * 19200 + 3200) * 4, stream>>>(mu_out,
        cl0_w, cl0_b, cl1_w, cl1_b, spl_w, spl_b, cls_out, leaves);
    length_kernel<<<1, 512, 3200 * 4, stream>>>(mu_out, ld0_w, ld0_b, ld1_w, ld1_b, len_out);

    // 7) leaf contributions to decoder gates (WMMA, transposed f32 output)
    {
        int tiles = (LROWS / 16) * (NTG / 2);           // 2432 waves
        int blocks = (tiles * 32 + 255) / 256;
        gemm_wmma<false, false, true><<<blocks, 256, 0, stream>>>(
            leaves, nullptr, HLD, 0, LROWS / 16, 10, pkLeafW, NTG, nullptr, leafG, nullptr, LROWS);
    }
    // 8) decoder recurrence
    lstm_seq<true, false><<<1, 1024, 0, stream>>>(preDec, leafG, pkDecHh, nullptr, hsD, TT);

    // 9) output projection hs_d @ dec_out_w + b (WMMA, bf16 A, row-major C)
    {
        int tiles = (ROWS / 16) * 12;                   // 98304 waves
        int blocks = (tiles * 32 + 255) / 256;
        gemm_wmma<true, false, false><<<blocks, 256, 0, stream>>>(
            nullptr, hsD, HLD, 0, ROWS / 16, 10, pkOutW, 24, dec_out_b, dout, nullptr, 384);
    }
}
